// lidar_feature_extractor_25761213842034
// MI455X (gfx1250) — matrix operand order repaired and checked by reading, not compiled, not run on hardware
//
#include <hip/hip_runtime.h>

typedef __bf16 bf16;
typedef __attribute__((ext_vector_type(16))) __bf16 v16bf;
typedef __attribute__((ext_vector_type(8)))  float  v8f;

#define KNN 16

// ---------------------------------------------------------------------------
// WMMA fragment loaders (CDNA5 v_wmma_f32_16x16x32_bf16; contiguous k order, same in A and B)
// ---------------------------------------------------------------------------
// A: 16x32 bf16 tile from LDS (row stride `as` elements). lane l: M = l&15,
// element i -> K = 16*(l>>4)+i.
__device__ __forceinline__ v16bf load_a_frag(const bf16* a, int as, int k0, int lane) {
  int m = lane & 15;
  int koff = (lane >> 4) * 16;
  const bf16* p0 = a + m * as + k0 + koff;
  v16bf r;
#pragma unroll
  for (int i = 0; i < 16; ++i) r[i] = p0[i];
  return r;
}

// B: 32x16 bf16 from a pre-transposed weight WT[Nc][Kd] in global memory.
// lane l: N = n0 + (l&15), element e -> K = k0 + 16*(l>>4) + e  (16 contiguous
// bf16 per lane -> vectorizable 32B global load).
__device__ __forceinline__ v16bf load_b_frag(const bf16* wt, int kd, int n0, int k0, int lane) {
  const bf16* p = wt + (size_t)(n0 + (lane & 15)) * kd + k0 + (lane >> 4) * 16;
  v16bf r;
#pragma unroll
  for (int i = 0; i < 16; ++i) r[i] = p[i];
  return r;
}

// ---------------------------------------------------------------------------
// Wave-cooperative GEMM: C[16 x 512] = act(A[16 x kd] @ WT^T + bias)
// 256-thread block = 8 waves; wave w covers output columns [64w, 64w+64).
// A is a bf16 LDS tile; WT is bf16 [512][kd] in global (L2 resident).
// Epilogue can write f32 (stride osF) and/or bf16 (stride osB).
// ---------------------------------------------------------------------------
__device__ __forceinline__ void wave_gemm_16x512(
    const bf16* sA, int as, const bf16* __restrict__ WT, int kd,
    const float* __restrict__ bias,
    float* outF, int osF, bf16* outB, int osB, int relu)
{
  int lane = threadIdx.x & 31;
  int wave = threadIdx.x >> 5;
  int n0 = wave * 64;
  v8f acc[4] = {};
  for (int k0 = 0; k0 < kd; k0 += 32) {
    v16bf a = load_a_frag(sA, as, k0, lane);
#pragma unroll
    for (int t = 0; t < 4; ++t) {
      v16bf b = load_b_frag(WT, kd, n0 + t * 16, k0, lane);
      acc[t] = __builtin_amdgcn_wmma_f32_16x16x32_bf16(
          false, a, false, b, (short)0, acc[t], false, false);
    }
  }
  int colb = lane & 15;
  int rowb = (lane >> 4) * 8;
#pragma unroll
  for (int t = 0; t < 4; ++t) {
    int col = n0 + t * 16 + colb;
    float bs = bias ? bias[col] : 0.0f;
#pragma unroll
    for (int r = 0; r < 8; ++r) {
      float v = acc[t][r] + bs;
      if (relu) v = fmaxf(v, 0.0f);
      if (outF) outF[(size_t)(rowb + r) * osF + col] = v;
      if (outB) outB[(size_t)(rowb + r) * osB + col] = (bf16)v;
    }
  }
}

// ---------------------------------------------------------------------------
// Standalone GEMM: C[M x 512] = act(A[M x Kd] @ WT^T + bias). grid.x = M/16.
// ---------------------------------------------------------------------------
__global__ __launch_bounds__(256) void gemm_bf16w_nc512(
    const float* __restrict__ A, const bf16* __restrict__ WT,
    const float* __restrict__ bias, float* __restrict__ C, int Kd, int relu)
{
  extern __shared__ char smem[];
  bf16* sA = (bf16*)smem;
  int as = Kd + 8;
  int tid = threadIdx.x;
  size_t rowbase = (size_t)blockIdx.x * 16;
  for (int e = tid; e < 16 * Kd; e += 256) {
    int r = e / Kd, c = e - r * Kd;
    sA[r * as + c] = (bf16)A[(rowbase + r) * Kd + c];
  }
  __syncthreads();
  wave_gemm_16x512(sA, as, WT, Kd, bias, C + rowbase * 512, 512, nullptr, 0, relu);
}

// Transpose + convert weight W[rows][cols] f32 -> WT[cols][rows] bf16.
__global__ void wtrans_bf16_kernel(const float* __restrict__ W, bf16* __restrict__ WT,
                                   int rows, int cols) {
  int i = blockIdx.x * 256 + threadIdx.x;
  if (i < rows * cols) {
    int r = i / cols, c = i - r * cols;
    WT[(size_t)c * rows + r] = (bf16)W[i];
  }
}

// ---------------------------------------------------------------------------
// feats = relu(x@w1 + b1)@w2 + b2   (3->32->32), one thread per point.
// ---------------------------------------------------------------------------
__global__ void feats_mlp_kernel(const float* __restrict__ X,
                                 const float* __restrict__ w1, const float* __restrict__ b1,
                                 const float* __restrict__ w2, const float* __restrict__ b2,
                                 float* __restrict__ F, int total) {
  int i = blockIdx.x * 256 + threadIdx.x;
  if (i >= total) return;
  float x0 = X[(size_t)i * 3], x1 = X[(size_t)i * 3 + 1], x2 = X[(size_t)i * 3 + 2];
  float h[32];
#pragma unroll
  for (int j = 0; j < 32; ++j)
    h[j] = fmaxf(b1[j] + x0 * w1[j] + x1 * w1[32 + j] + x2 * w1[64 + j], 0.0f);
#pragma unroll
  for (int j = 0; j < 32; ++j) {
    float a = b2[j];
#pragma unroll
    for (int k = 0; k < 32; ++k) a += h[k] * w2[k * 32 + j];
    F[(size_t)i * 32 + j] = a;
  }
}

// ---------------------------------------------------------------------------
// 16-NN by squared distance; register insertion sort, ties -> lower index.
// One thread per query (i in [0, B*Nq)).
// ---------------------------------------------------------------------------
__global__ void knn_kernel(const float* __restrict__ Q, const float* __restrict__ R,
                           int* __restrict__ IDX, int Nq, int Nr, int total) {
  int i = blockIdx.x * 256 + threadIdx.x;
  if (i >= total) return;
  int b = i / Nq;
  float qx = Q[(size_t)i * 3], qy = Q[(size_t)i * 3 + 1], qz = Q[(size_t)i * 3 + 2];
  const float* rb = R + (size_t)b * Nr * 3;
  float bd[KNN]; int bi[KNN];
#pragma unroll
  for (int s = 0; s < KNN; ++s) { bd[s] = 3.0e38f; bi[s] = 0; }
  for (int j = 0; j < Nr; ++j) {
    float dx = qx - rb[j * 3], dy = qy - rb[j * 3 + 1], dz = qz - rb[j * 3 + 2];
    float cd = dx * dx + dy * dy + dz * dz;
    int ci = j;
#pragma unroll
    for (int s = 0; s < KNN; ++s) {
      bool sw = cd < bd[s];
      float tf = bd[s]; int ti = bi[s];
      if (sw) { bd[s] = cd; bi[s] = ci; cd = tf; ci = ti; }
    }
  }
#pragma unroll
  for (int s = 0; s < KNN; ++s) IDX[(size_t)i * KNN + s] = bi[s];
}

// ---------------------------------------------------------------------------
// Farthest point sampling: one 1024-thread block per batch, serial 256 steps.
// Emits the incoming `far` each step (matches the reference scan semantics).
// ---------------------------------------------------------------------------
__global__ __launch_bounds__(1024) void fps_kernel(
    const float* __restrict__ XYZ, int* __restrict__ FID, float* __restrict__ NXYZ,
    int N, int npoint) {
  __shared__ float s_d[1024];
  __shared__ int   s_i[1024];
  __shared__ float s_c[3];
  int b = blockIdx.x, t = threadIdx.x;
  const float* xb = XYZ + (size_t)b * N * 3;
  float px = xb[t * 3], py = xb[t * 3 + 1], pz = xb[t * 3 + 2];
  float dist = 1.0e10f;
  int far = 0;
  for (int it = 0; it < npoint; ++it) {
    if (t == 0) {
      FID[b * npoint + it] = far;
      s_c[0] = xb[far * 3]; s_c[1] = xb[far * 3 + 1]; s_c[2] = xb[far * 3 + 2];
      NXYZ[(size_t)(b * npoint + it) * 3 + 0] = s_c[0];
      NXYZ[(size_t)(b * npoint + it) * 3 + 1] = s_c[1];
      NXYZ[(size_t)(b * npoint + it) * 3 + 2] = s_c[2];
    }
    __syncthreads();
    float dx = px - s_c[0], dy = py - s_c[1], dz = pz - s_c[2];
    dist = fminf(dist, dx * dx + dy * dy + dz * dz);
    s_d[t] = dist; s_i[t] = t;
    __syncthreads();
    for (int o = 512; o > 0; o >>= 1) {
      if (t < o) {
        if (s_d[t + o] > s_d[t] || (s_d[t + o] == s_d[t] && s_i[t + o] < s_i[t])) {
          s_d[t] = s_d[t + o]; s_i[t] = s_i[t + o];
        }
      }
      __syncthreads();
    }
    far = s_i[0];
    __syncthreads();
  }
}

// ---------------------------------------------------------------------------
// Transition-down grouping + 2x (conv + BN + relu) + max over K.
// One 64-thread block per (b, q); thread = output channel.
// ---------------------------------------------------------------------------
__global__ __launch_bounds__(64) void td_kernel(
    const float* __restrict__ XYZ, const float* __restrict__ NXYZ,
    const int* __restrict__ IDX, const float* __restrict__ PTS,
    const float* __restrict__ c1w, const float* __restrict__ c1b,
    const float* __restrict__ bn1g, const float* __restrict__ bn1b,
    const float* __restrict__ bn1m, const float* __restrict__ bn1v,
    const float* __restrict__ c2w, const float* __restrict__ c2b,
    const float* __restrict__ bn2g, const float* __restrict__ bn2b,
    const float* __restrict__ bn2m, const float* __restrict__ bn2v,
    float* __restrict__ OUT, int N, int M) {
  __shared__ float s_f[35];
  __shared__ float s_h1[64];
  __shared__ int   s_idx[16];
  __shared__ float s_c[3];
  int bq = blockIdx.x;
  int b = bq / M;
  int c = threadIdx.x;
  if (c < 16) s_idx[c] = IDX[(size_t)bq * 16 + c];
  if (c < 3)  s_c[c] = NXYZ[(size_t)bq * 3 + c];
  __syncthreads();
  float sc1 = bn1g[c] * rsqrtf(bn1v[c] + 1e-5f);
  float sh1 = bn1b[c] - bn1m[c] * sc1;
  float sc2 = bn2g[c] * rsqrtf(bn2v[c] + 1e-5f);
  float sh2 = bn2b[c] - bn2m[c] * sc2;
  float mx = -3.0e38f;
  for (int k = 0; k < 16; ++k) {
    int id = s_idx[k];
    if (c < 3)             s_f[c] = XYZ[((size_t)b * N + id) * 3 + c] - s_c[c];
    if (c >= 3 && c < 35)  s_f[c] = PTS[((size_t)b * N + id) * 32 + (c - 3)];
    __syncthreads();
    float a1 = c1b[c];
    for (int j = 0; j < 35; ++j) a1 += s_f[j] * c1w[j * 64 + c];
    s_h1[c] = fmaxf(a1 * sc1 + sh1, 0.0f);
    __syncthreads();
    float a2 = c2b[c];
    for (int j = 0; j < 64; ++j) a2 += s_h1[j] * c2w[j * 64 + c];
    mx = fmaxf(mx, fmaxf(a2 * sc2 + sh2, 0.0f));
    __syncthreads();
  }
  OUT[(size_t)bq * 64 + c] = mx;
}

// ---------------------------------------------------------------------------
// Fused point-transformer attention: one workgroup (256 thr = 8 waves) per
// (b, n). Keeps the 16x512 tiles in LDS; three WMMA GEMMs against L2-resident
// bf16 weights; softmax over K per channel; weighted sum; fc2 + residual.
// ---------------------------------------------------------------------------
__global__ __launch_bounds__(256) void attn_block_kernel(
    const float* __restrict__ XYZ, const int* __restrict__ IDX,
    const float* __restrict__ Qm, const float* __restrict__ Km, const float* __restrict__ Vm,
    const float* __restrict__ dw1, const float* __restrict__ db1,
    const bf16* __restrict__ dw2T, const float* __restrict__ db2,
    const bf16* __restrict__ gw1T, const float* __restrict__ gb1,
    const bf16* __restrict__ gw2T, const float* __restrict__ gb2,
    const float* __restrict__ fc2w, const float* __restrict__ fc2b,
    const float* __restrict__ PRE, float* __restrict__ OUT,
    int N, int din) {
  extern __shared__ char smem[];
  char* p = smem;
  float* s_pos  = (float*)p; p += 16 * 512 * 4;   // 32768
  float* s_attn = (float*)p; p += 16 * 512 * 4;   // 32768
  bf16*  s_a    = (bf16*)p;  p += 16 * 520 * 2;   // 16640
  bf16*  s_h    = (bf16*)p;  p += 16 * 520 * 2;   // 16640
  float* s_q    = (float*)p; p += 512 * 4;
  float* s_res  = (float*)p; p += 512 * 4;
  int*   s_idx  = (int*)p;   p += 64;
  float* s_dif  = (float*)p;                      // 48 + 3 floats
  float* s_c    = s_dif + 48;
  const int AH = 520;

  int tid = threadIdx.x;
  int bn_ = blockIdx.x;                 // b*N + n
  size_t rowq = (size_t)bn_ * 512;
  size_t base = (size_t)(bn_ / N) * N;  // b*N

  if (tid < 16) s_idx[tid] = IDX[(size_t)bn_ * 16 + tid];
  if (tid < 3)  s_c[tid] = XYZ[(size_t)bn_ * 3 + tid];
  for (int f = tid; f < 512; f += 256) s_q[f] = Qm[rowq + f];
  __syncthreads();
  if (tid < 48) {
    int k = tid / 3, d = tid - k * 3;
    s_dif[tid] = s_c[d] - XYZ[(base + s_idx[k]) * 3 + d];   // center - neighbor
  }
  __syncthreads();

  // pos layer 1: relu(diff @ d_w1 + d_b1) -> bf16 LDS
  for (int e = tid; e < 16 * 512; e += 256) {
    int k = e >> 9, f = e & 511;
    float v = db1[f] + s_dif[k * 3] * dw1[f] + s_dif[k * 3 + 1] * dw1[512 + f]
            + s_dif[k * 3 + 2] * dw1[1024 + f];
    s_h[k * AH + f] = (bf16)fmaxf(v, 0.0f);
  }
  __syncthreads();
  // pos = h @ d_w2 + d_b2   (WMMA)
  wave_gemm_16x512(s_h, AH, dw2T, 512, db2, s_pos, 512, nullptr, 0, 0);
  __syncthreads();
  // attn_in = q - k + pos  -> bf16
  for (int e = tid; e < 16 * 512; e += 256) {
    int k = e >> 9, f = e & 511;
    float kv = Km[(base + s_idx[k]) * 512 + f];
    s_a[k * AH + f] = (bf16)(s_q[f] - kv + s_pos[k * 512 + f]);
  }
  __syncthreads();
  // h = relu(attn_in @ g_w1 + g_b1)   (WMMA)
  wave_gemm_16x512(s_a, AH, gw1T, 512, gb1, nullptr, 0, s_h, AH, 1);
  __syncthreads();
  // attn = h @ g_w2 + g_b2   (WMMA)
  wave_gemm_16x512(s_h, AH, gw2T, 512, gb2, s_attn, 512, nullptr, 0, 0);
  __syncthreads();

  // softmax over K (axis=-2) per channel, scaled by 1/sqrt(512)
  const float invs = 0.044194173824159216f;
  for (int f = tid; f < 512; f += 256) {
    float mx = -3.0e38f;
#pragma unroll
    for (int k = 0; k < 16; ++k) mx = fmaxf(mx, s_attn[k * 512 + f]);
    float ex[16];
    float sm = 0.0f;
#pragma unroll
    for (int k = 0; k < 16; ++k) {
      float e2 = __expf((s_attn[k * 512 + f] - mx) * invs);
      ex[k] = e2; sm += e2;
    }
    float inv = 1.0f / sm;
#pragma unroll
    for (int k = 0; k < 16; ++k) s_attn[k * 512 + f] = ex[k] * inv;
  }
  __syncthreads();

  // res = sum_k attn * (v + pos)
  for (int f = tid; f < 512; f += 256) {
    float r = 0.0f;
#pragma unroll
    for (int k = 0; k < 16; ++k)
      r += s_attn[k * 512 + f] * (Vm[(base + s_idx[k]) * 512 + f] + s_pos[k * 512 + f]);
    s_res[f] = r;
  }
  __syncthreads();

  // out = res @ fc2_w + fc2_b + pre
  for (int c = tid; c < din; c += 256) {
    float a = fc2b[c];
    for (int f = 0; f < 512; ++f) a += s_res[f] * fc2w[f * din + c];
    OUT[(size_t)bn_ * din + c] = a + PRE[(size_t)bn_ * din + c];
  }
}

// ---------------------------------------------------------------------------
// Host orchestration
// ---------------------------------------------------------------------------
extern "C" void kernel_launch(void* const* d_in, const int* in_sizes, int n_in,
                              void* d_out, int out_size, void* d_ws, size_t ws_size,
                              hipStream_t stream) {
  (void)in_sizes; (void)n_in; (void)out_size; (void)ws_size;
  const int B = 16, N1 = 1024, N2 = 256;

  // setup_inputs() dict insertion order:
  // 0:x 1:fc1_w1 2:fc1_b1 3:fc1_w2 4:fc1_b2
  // t1 @5:  fc1_w fc1_b wq wk wv d_w1 d_b1 d_w2 d_b2 g_w1 g_b1 g_w2 g_b2 fc2_w fc2_b
  // td @20: c1_w c1_b bn1_g bn1_b bn1_m bn1_v c2_w c2_b bn2_g bn2_b bn2_m bn2_v
  // t2 @32: (same 15 as t1)
  auto P = [&](int i) { return (const float*)d_in[i]; };
  const float* X = P(0);
  const int T1 = 5, TD = 20, T2 = 32;

  // workspace carve (256B aligned)
  char* w = (char*)d_ws;
  auto alloc = [&](size_t bytes) -> char* {
    char* q = w; w += (bytes + 255) & ~(size_t)255; return q;
  };
  float* feats  = (float*)alloc((size_t)B * N1 * 32 * 4);
  int*   knn1   = (int*)alloc((size_t)B * N1 * 16 * 4);
  float* xproj  = (float*)alloc((size_t)B * N1 * 512 * 4);  // reused by t2
  float* qb     = (float*)alloc((size_t)B * N1 * 512 * 4);
  float* kb     = (float*)alloc((size_t)B * N1 * 512 * 4);
  float* vb     = (float*)alloc((size_t)B * N1 * 512 * 4);
  float* pts1   = (float*)alloc((size_t)B * N1 * 32 * 4);
  int*   fid    = (int*)alloc((size_t)B * N2 * 4);
  float* nxyz   = (float*)alloc((size_t)B * N2 * 3 * 4);
  int*   knn_td = (int*)alloc((size_t)B * N2 * 16 * 4);
  float* pts_td = (float*)alloc((size_t)B * N2 * 64 * 4);
  int*   knn2   = (int*)alloc((size_t)B * N2 * 16 * 4);
  bf16* fc1T1 = (bf16*)alloc((size_t)32 * 512 * 2);
  bf16* wqT1  = (bf16*)alloc((size_t)512 * 512 * 2);
  bf16* wkT1  = (bf16*)alloc((size_t)512 * 512 * 2);
  bf16* wvT1  = (bf16*)alloc((size_t)512 * 512 * 2);
  bf16* dw2T1 = (bf16*)alloc((size_t)512 * 512 * 2);
  bf16* gw1T1 = (bf16*)alloc((size_t)512 * 512 * 2);
  bf16* gw2T1 = (bf16*)alloc((size_t)512 * 512 * 2);
  bf16* fc1T2 = (bf16*)alloc((size_t)64 * 512 * 2);
  bf16* wqT2  = (bf16*)alloc((size_t)512 * 512 * 2);
  bf16* wkT2  = (bf16*)alloc((size_t)512 * 512 * 2);
  bf16* wvT2  = (bf16*)alloc((size_t)512 * 512 * 2);
  bf16* dw2T2 = (bf16*)alloc((size_t)512 * 512 * 2);
  bf16* gw1T2 = (bf16*)alloc((size_t)512 * 512 * 2);
  bf16* gw2T2 = (bf16*)alloc((size_t)512 * 512 * 2);

  auto wt = [&](const float* src, bf16* dst, int rows, int cols) {
    int n = rows * cols;
    wtrans_bf16_kernel<<<(n + 255) / 256, 256, 0, stream>>>(src, dst, rows, cols);
  };
  wt(P(T1 + 0), fc1T1, 32, 512);
  wt(P(T1 + 2), wqT1, 512, 512);  wt(P(T1 + 3), wkT1, 512, 512);
  wt(P(T1 + 4), wvT1, 512, 512);  wt(P(T1 + 7), dw2T1, 512, 512);
  wt(P(T1 + 9), gw1T1, 512, 512); wt(P(T1 + 11), gw2T1, 512, 512);
  wt(P(T2 + 0), fc1T2, 64, 512);
  wt(P(T2 + 2), wqT2, 512, 512);  wt(P(T2 + 3), wkT2, 512, 512);
  wt(P(T2 + 4), wvT2, 512, 512);  wt(P(T2 + 7), dw2T2, 512, 512);
  wt(P(T2 + 9), gw1T2, 512, 512); wt(P(T2 + 11), gw2T2, 512, 512);

  // feats + knn(t1)
  feats_mlp_kernel<<<(B * N1 + 255) / 256, 256, 0, stream>>>(
      X, P(1), P(2), P(3), P(4), feats, B * N1);
  knn_kernel<<<(B * N1 + 255) / 256, 256, 0, stream>>>(X, X, knn1, N1, N1, B * N1);

  // t1 projections (WMMA GEMMs)
  size_t sm32 = 16 * (32 + 8) * 2, sm64 = 16 * (64 + 8) * 2, sm512 = 16 * (512 + 8) * 2;
  gemm_bf16w_nc512<<<B * N1 / 16, 256, sm32, stream>>>(feats, fc1T1, P(T1 + 1), xproj, 32, 0);
  gemm_bf16w_nc512<<<B * N1 / 16, 256, sm512, stream>>>(xproj, wqT1, nullptr, qb, 512, 0);
  gemm_bf16w_nc512<<<B * N1 / 16, 256, sm512, stream>>>(xproj, wkT1, nullptr, kb, 512, 0);
  gemm_bf16w_nc512<<<B * N1 / 16, 256, sm512, stream>>>(xproj, wvT1, nullptr, vb, 512, 0);

  // t1 fused attention
  size_t smAttn = 103424;
  attn_block_kernel<<<B * N1, 256, smAttn, stream>>>(
      X, knn1, qb, kb, vb,
      P(T1 + 5), P(T1 + 6), dw2T1, P(T1 + 8), gw1T1, P(T1 + 10), gw2T1, P(T1 + 12),
      P(T1 + 13), P(T1 + 14), feats, pts1, N1, 32);

  // transition down
  fps_kernel<<<B, 1024, 0, stream>>>(X, fid, nxyz, N1, N2);
  knn_kernel<<<(B * N2 + 255) / 256, 256, 0, stream>>>(nxyz, X, knn_td, N2, N1, B * N2);
  td_kernel<<<B * N2, 64, 0, stream>>>(
      X, nxyz, knn_td, pts1,
      P(TD + 0), P(TD + 1), P(TD + 2), P(TD + 3), P(TD + 4), P(TD + 5),
      P(TD + 6), P(TD + 7), P(TD + 8), P(TD + 9), P(TD + 10), P(TD + 11),
      pts_td, N1, N2);

  // t2
  knn_kernel<<<(B * N2 + 255) / 256, 256, 0, stream>>>(nxyz, nxyz, knn2, N2, N2, B * N2);
  gemm_bf16w_nc512<<<B * N2 / 16, 256, sm64, stream>>>(pts_td, fc1T2, P(T2 + 1), xproj, 64, 0);
  gemm_bf16w_nc512<<<B * N2 / 16, 256, sm512, stream>>>(xproj, wqT2, nullptr, qb, 512, 0);
  gemm_bf16w_nc512<<<B * N2 / 16, 256, sm512, stream>>>(xproj, wkT2, nullptr, kb, 512, 0);
  gemm_bf16w_nc512<<<B * N2 / 16, 256, sm512, stream>>>(xproj, wvT2, nullptr, vb, 512, 0);
  attn_block_kernel<<<B * N2, 256, smAttn, stream>>>(
      nxyz, knn2, qb, kb, vb,
      P(T2 + 5), P(T2 + 6), dw2T2, P(T2 + 8), gw1T2, P(T2 + 10), gw2T2, P(T2 + 12),
      P(T2 + 13), P(T2 + 14), pts_td, (float*)d_out, N2, 64);
}
